// ReformerAttention_90675349553512
// MI455X (gfx1250) — matrix operand order repaired and checked by reading, not compiled, not run on hardware
//
#include <hip/hip_runtime.h>
#include <hip/hip_bf16.h>

typedef __attribute__((ext_vector_type(16))) _Float16 v16h;
typedef __attribute__((ext_vector_type(8)))  float    v8f;

#define SEQ     4096
#define NHASH   8
#define NCHUNK  512          // N_HASHES * n_buckets
#define NITEMS  32768        // NHASH * SEQ
#define NBUCKET 512          // bucket ids incl. hash offset
#define NSEG    256          // 32768 / 128
#define SEGSZ   128
#define EDIM    64
#define NROWS   16           // b*h

// ---------------- phase 1: LSH hashing ----------------
__global__ void __launch_bounds__(256)
lsh_hash_kernel(const float* __restrict__ qk, const float* __restrict__ rot,
                unsigned* __restrict__ buckets) {
    int idx = blockIdx.x * 256 + threadIdx.x;       // (r, h, t)
    int t = idx & (SEQ - 1);
    int h = (idx >> 12) & 7;
    int r = idx >> 15;
    int b_ = r >> 3, h_ = r & 7;
    const float* q = qk + ((size_t)(b_ * SEQ + t) * NHASH + h_) * EDIM;
    float acc[32];
#pragma unroll
    for (int i = 0; i < 32; ++i) acc[i] = 0.f;
    for (int e = 0; e < EDIM; ++e) {
        float qe = q[e];
        const float* rp = rot + ((size_t)e * NHASH + h) * 32;
#pragma unroll
        for (int i = 0; i < 32; ++i) acc[i] = fmaf(qe, rp[i], acc[i]);
    }
    float best = -3.0e38f; int bi = 0;
#pragma unroll
    for (int i = 0; i < 32; ++i) if (acc[i] > best) { best = acc[i]; bi = i; }
#pragma unroll
    for (int i = 0; i < 32; ++i) if (-acc[i] > best) { best = -acc[i]; bi = 32 + i; }
    buckets[(size_t)r * NITEMS + h * SEQ + t] = (unsigned)(bi + h * 64);
}

// ---------------- phase 2a: per-segment histogram ----------------
__global__ void __launch_bounds__(SEGSZ)
hist_kernel(const unsigned* __restrict__ buckets, unsigned* __restrict__ hist) {
    int seg = blockIdx.x, r = blockIdx.y;
    int j = seg * SEGSZ + threadIdx.x;
    unsigned bk = buckets[(size_t)r * NITEMS + j];
    atomicAdd(&hist[(size_t)r * (NBUCKET * NSEG) + bk * NSEG + seg], 1u);
}

// ---------------- phase 2b: bucket-major exclusive scan (131072 per row) ----------------
__global__ void __launch_bounds__(1024)
scan_kernel(unsigned* __restrict__ hist) {
    __shared__ unsigned sd[1024];
    __shared__ unsigned carry;
    unsigned* h = hist + (size_t)blockIdx.x * (NBUCKET * NSEG);
    int tid = threadIdx.x;
    if (tid == 0) carry = 0;
    __syncthreads();
    for (int it = 0; it < (NBUCKET * NSEG) / 1024; ++it) {
        unsigned x = h[it * 1024 + tid];
        sd[tid] = x;
        __syncthreads();
        for (int ofs = 1; ofs < 1024; ofs <<= 1) {
            unsigned tv = (tid >= ofs) ? sd[tid - ofs] : 0u;
            __syncthreads();
            sd[tid] += tv;
            __syncthreads();
        }
        unsigned incl = sd[tid];
        unsigned c = carry;
        h[it * 1024 + tid] = c + incl - x;     // exclusive + carry
        __syncthreads();
        if (tid == 1023) carry = c + incl;
        __syncthreads();
    }
}

// ---------------- phase 2c: rank-stable scatter ----------------
__global__ void __launch_bounds__(SEGSZ)
scatter_kernel(const unsigned* __restrict__ buckets, const unsigned* __restrict__ offs,
               unsigned* __restrict__ sticker) {
    int seg = blockIdx.x, r = blockIdx.y, tid = threadIdx.x;
    __shared__ unsigned sb[SEGSZ];
    int j = seg * SEGSZ + tid;
    unsigned bk = buckets[(size_t)r * NITEMS + j];
    sb[tid] = bk;
    __syncthreads();
    unsigned rank = 0;
    for (int i = 0; i < tid; ++i) rank += (sb[i] == bk) ? 1u : 0u;
    unsigned pos = offs[(size_t)r * (NBUCKET * NSEG) + bk * NSEG + seg] + rank;
    sticker[(size_t)r * NITEMS + pos] = (unsigned)j;
}

// ---------------- WMMA fragment loaders (ISA 7.12.2 16-bit layouts) ----------------
__device__ __forceinline__ v16h load_a16(const _Float16* s, int ld, int m0, int k0,
                                         int ln, int hi) {
    // A 16x32: lane = hi*16+ln holds row m0+ln; K = k0 + hi*16 + g (contiguous)
    const _Float16* p = s + (size_t)(m0 + ln) * ld + k0 + hi * 16;
    v16h r;
#pragma unroll
    for (int g = 0; g < 16; ++g) r[g] = p[g];
    return r;
}
__device__ __forceinline__ v16h load_bT(const _Float16* s, int ld, int n0, int k0,
                                        int ln, int hi) {
    // B[k][n] = s[(n0+n)*ld + k0+k]  (keys row-major -> B = K^T): contiguous per lane
    const _Float16* p = s + (size_t)(n0 + ln) * ld + k0 + hi * 16;
    v16h r;
#pragma unroll
    for (int g = 0; g < 16; ++g) r[g] = p[g];
    return r;
}
__device__ __forceinline__ v16h load_bN(const _Float16* s, int ld, int k0, int n0,
                                        int ln, int hi) {
    // B[k][n] = s[(k0+k)*ld + n0+n]  (values row-major): strided per lane
    const _Float16* p = s + (size_t)(k0 + hi * 16) * ld + n0 + ln;
    v16h r;
#pragma unroll
    for (int g = 0; g < 16; ++g) r[g] = p[(size_t)g * ld];
    return r;
}

// ---------------- phase 3: bucketed attention (WMMA) ----------------
__global__ void __launch_bounds__(128)
attn_kernel(const float* __restrict__ qk, const float* __restrict__ v,
            const unsigned* __restrict__ sticker,
            _Float16* __restrict__ o_all, float* __restrict__ lse_all) {
    int c = blockIdx.x;          // chunk
    int r = blockIdx.y;          // (b*h) row
    int b_ = r >> 3, h_ = r & 7;

    __shared__ _Float16 qA[64 * 64];     // bq (f16, row-major)
    __shared__ _Float16 kB[128 * 64];    // normalized keys (cur + look-back)
    __shared__ _Float16 vB[128 * 64];    // values
    __shared__ _Float16 pA[64 * 128];    // softmax probs
    __shared__ unsigned tIdx[128];       // token ids (st) for keys
    __shared__ unsigned itemIdx[64];     // sorted item id (h*SEQ + t) for queries

    int tid = threadIdx.x;
    int wave = tid >> 5, lane = tid & 31;
    int hi = lane >> 4, ln = lane & 15;

    const unsigned* srow = sticker + (size_t)r * NITEMS;
    {   // gather indices: 64 current (queries+keys) + 64 from previous chunk (keys)
        int cc = (tid < 64) ? c : (c + NCHUNK - 1) % NCHUNK;
        unsigned item = srow[cc * 64 + (tid & 63)];
        tIdx[tid] = item & (SEQ - 1);
        if (tid < 64) itemIdx[tid] = item;
    }
    __syncthreads();

    // gather rows from global, normalize keys, convert f32 -> f16
    for (int row = wave; row < 128; row += 4) {
        unsigned t = tIdx[row];
        size_t base = ((size_t)(b_ * SEQ + (int)t) * NHASH + h_) * EDIM;
        float2 x  = *(const float2*)(qk + base + lane * 2);
        float2 vv = *(const float2*)(v  + base + lane * 2);
        float ss = x.x * x.x + x.y * x.y;
#pragma unroll
        for (int m = 16; m > 0; m >>= 1) ss += __shfl_xor(ss, m, 32);
        float inv = rsqrtf(ss < 1e-24f ? 1e-24f : ss);   // == 1/max(||q||,1e-12)
        kB[row * 64 + lane * 2]     = (_Float16)(x.x * inv);
        kB[row * 64 + lane * 2 + 1] = (_Float16)(x.y * inv);
        vB[row * 64 + lane * 2]     = (_Float16)vv.x;
        vB[row * 64 + lane * 2 + 1] = (_Float16)vv.y;
        if (row < 64) {
            qA[row * 64 + lane * 2]     = (_Float16)x.x;
            qA[row * 64 + lane * 2 + 1] = (_Float16)x.y;
        }
    }
    __syncthreads();

    // ---- dots = bq @ bk^T : wave owns 16 query rows x all 128 key cols ----
    int m0 = wave * 16;
    v16h a0 = load_a16(qA, 64, m0, 0, ln, hi);
    v16h a1 = load_a16(qA, 64, m0, 32, ln, hi);
    v8f acc[8];
#pragma unroll
    for (int ct = 0; ct < 8; ++ct) {
        v16h b0 = load_bT(kB, 64, ct * 16, 0, ln, hi);
        v16h b1 = load_bT(kB, 64, ct * 16, 32, ln, hi);
        v8f d = {};
        d = __builtin_amdgcn_wmma_f32_16x16x32_f16(false, a0, false, b0, (short)0, d, false, false);
        d = __builtin_amdgcn_wmma_f32_16x16x32_f16(false, a1, false, b1, (short)0, d, false, false);
        acc[ct] = d;
    }

    // ---- self-mask + row softmax in C layout (row = VGPR across a 16-lane half) ----
    unsigned tq[8];
#pragma unroll
    for (int g = 0; g < 8; ++g) tq[g] = tIdx[m0 + g + hi * 8];
    float rmax[8];
#pragma unroll
    for (int g = 0; g < 8; ++g) rmax[g] = -3.0e38f;
#pragma unroll
    for (int ct = 0; ct < 8; ++ct) {
        unsigned tk = tIdx[ct * 16 + ln];
#pragma unroll
        for (int g = 0; g < 8; ++g) {
            float d = acc[ct][g];
            if (tq[g] == tk) d = -50000.0f;       // TOKEN_SELF_ATTN_VALUE
            acc[ct][g] = d;
            rmax[g] = fmaxf(rmax[g], d);
        }
    }
#pragma unroll
    for (int g = 0; g < 8; ++g)
#pragma unroll
        for (int m = 8; m > 0; m >>= 1)           // xor 8/4/2/1 stays in 16-lane group
            rmax[g] = fmaxf(rmax[g], __shfl_xor(rmax[g], m, 32));
    float rsum[8];
#pragma unroll
    for (int g = 0; g < 8; ++g) rsum[g] = 0.f;
#pragma unroll
    for (int ct = 0; ct < 8; ++ct)
#pragma unroll
        for (int g = 0; g < 8; ++g) {
            float e = __expf(acc[ct][g] - rmax[g]);
            acc[ct][g] = e;
            rsum[g] += e;
        }
#pragma unroll
    for (int g = 0; g < 8; ++g)
#pragma unroll
        for (int m = 8; m > 0; m >>= 1) rsum[g] += __shfl_xor(rsum[g], m, 32);
#pragma unroll
    for (int ct = 0; ct < 8; ++ct)
#pragma unroll
        for (int g = 0; g < 8; ++g)
            pA[(size_t)(m0 + g + hi * 8) * 128 + ct * 16 + ln] =
                (_Float16)(acc[ct][g] / rsum[g]);
    if (ln == 0) {
#pragma unroll
        for (int g = 0; g < 8; ++g) {
            int i = m0 + g + hi * 8;
            lse_all[(size_t)r * NITEMS + itemIdx[i]] = rmax[g] + __logf(rsum[g]);
        }
    }
    __syncthreads();

    // ---- bo = p @ bv : wave owns 16 rows x 64 e-cols, K = 128 ----
    v8f oacc[4];
#pragma unroll
    for (int nt = 0; nt < 4; ++nt) { v8f z = {}; oacc[nt] = z; }
#pragma unroll
    for (int k0 = 0; k0 < 128; k0 += 32) {
        v16h a = load_a16(pA, 128, m0, k0, ln, hi);
#pragma unroll
        for (int nt = 0; nt < 4; ++nt) {
            v16h bb = load_bN(vB, 64, k0, nt * 16, ln, hi);
            oacc[nt] = __builtin_amdgcn_wmma_f32_16x16x32_f16(false, a, false, bb,
                                                              (short)0, oacc[nt], false, false);
        }
    }
    // scatter to per-(hash,token) slot: item = h*SEQ + t
#pragma unroll
    for (int nt = 0; nt < 4; ++nt)
#pragma unroll
        for (int g = 0; g < 8; ++g) {
            int i = m0 + g + hi * 8;
            unsigned item = itemIdx[i];
            o_all[((size_t)r * NITEMS + item) * EDIM + nt * 16 + ln] = (_Float16)oacc[nt][g];
        }
}

// ---------------- phase 4: combine across hash rounds ----------------
__global__ void __launch_bounds__(256)
combine_kernel(const _Float16* __restrict__ o_all, const float* __restrict__ lse_all,
               float* __restrict__ out) {
    int gid = blockIdx.x * 256 + threadIdx.x;   // ((r*SEQ + t)*EDIM + e)
    int e = gid & 63;
    int t = (gid >> 6) & (SEQ - 1);
    int r = gid >> 18;
    const float* L = lse_all + (size_t)r * NITEMS + t;
    float l[8], m = -3.0e38f;
#pragma unroll
    for (int h = 0; h < 8; ++h) { l[h] = L[(size_t)h * SEQ]; m = fmaxf(m, l[h]); }
    float s = 0.f;
#pragma unroll
    for (int h = 0; h < 8; ++h) { l[h] = __expf(l[h] - m); s += l[h]; }
    float o = 0.f;
#pragma unroll
    for (int h = 0; h < 8; ++h)
        o += l[h] * (float)o_all[((size_t)r * NITEMS + h * SEQ + t) * EDIM + e];
    out[gid] = o / s;
}

extern "C" void kernel_launch(void* const* d_in, const int* in_sizes, int n_in,
                              void* d_out, int out_size, void* d_ws, size_t ws_size,
                              hipStream_t stream) {
    const float* qk  = (const float*)d_in[0];
    // d_in[1] ("k") is unused by the reference (shared-QK attention)
    const float* v   = (const float*)d_in[2];
    const float* rot = (const float*)d_in[3];
    float* out = (float*)d_out;

    char* ws = (char*)d_ws;
    unsigned* buckets = (unsigned*)(ws);                         //  2 MB
    unsigned* hist    = (unsigned*)(ws + ((size_t)2  << 20));    //  8 MB (hist -> offsets in place)
    unsigned* sticker = (unsigned*)(ws + ((size_t)10 << 20));    //  2 MB
    float*    lse_all = (float*)   (ws + ((size_t)12 << 20));    //  2 MB
    _Float16* o_all   = (_Float16*)(ws + ((size_t)14 << 20));    // 64 MB

    hipMemsetAsync(hist, 0, (size_t)NROWS * NBUCKET * NSEG * sizeof(unsigned), stream);
    lsh_hash_kernel<<<(NROWS * NHASH * SEQ) / 256, 256, 0, stream>>>(qk, rot, buckets);
    hist_kernel<<<dim3(NSEG, NROWS), SEGSZ, 0, stream>>>(buckets, hist);
    scan_kernel<<<NROWS, 1024, 0, stream>>>(hist);
    scatter_kernel<<<dim3(NSEG, NROWS), SEGSZ, 0, stream>>>(buckets, hist, sticker);
    attn_kernel<<<dim3(NCHUNK, NROWS), 128, 0, stream>>>(qk, v, sticker, o_all, lse_all);
    combine_kernel<<<(NROWS * SEQ * EDIM) / 256, 256, 0, stream>>>(o_all, lse_all, out);
}
